// GNN_45226005626987
// MI455X (gfx1250) — compile-verified, matrix operand order checked by reading, not run on hardware
//
#include <hip/hip_runtime.h>
#include <hip/hip_bf16.h>

typedef __attribute__((ext_vector_type(16))) _Float16 v16h;
typedef __attribute__((ext_vector_type(8)))  _Float16 v8h;
typedef __attribute__((ext_vector_type(8)))  float    v8f;

#define N_NODES 100000
#define N_EDGES 1600000
#define D_IN    50
#define HIDDEN  128
#define NCLS    8
#define MTILES  (N_NODES / 16)   // 6250 exactly

// ---- workspace layout (bytes, 256-aligned) ----
#define OFF_DEG   ((size_t)0)          // 100000 f32          = 400000
#define OFF_AGG1  ((size_t)400128)     // 100000*64 f32       = 25600000 (stride-64 padded agg)
#define OFF_H16   ((size_t)26000128)   // 100000*128 f16      = 25600000
#define OFF_Q     ((size_t)51600128)   // 100000*16 f32       = 6400000
#define OFF_AGGP  ((size_t)58000128)   // 100000*8 f32        = 3200000
#define OFF_WC1   ((size_t)61200128)   // 128*128 f16 (col-major) = 32768
#define OFF_WC2   ((size_t)61232896)   // 16*128 f16 (col-major)  = 4096
#define OFF_FC16  ((size_t)61236992)   // 100000*128 f16      = 25600000  (packed [agg/deg,x,0])
// total = 86,836,992 bytes

// Pack [W1_l ; W1_r ; 0] (K=128 x N=128) and [W2_l | W2_r] (K=128 x N=16)
// into column-major f16 so WMMA B fragments are contiguous 16B loads.
__global__ void prep_w_k(const float* __restrict__ W1l, const float* __restrict__ W1r,
                         const float* __restrict__ W2l, const float* __restrict__ W2r,
                         _Float16* __restrict__ wc1, _Float16* __restrict__ wc2) {
    int tid = blockIdx.x * blockDim.x + threadIdx.x;
    if (tid < 128 * 128) {
        int col = tid >> 7, k = tid & 127;
        float v = (k < D_IN)     ? W1l[k * HIDDEN + col]
                : (k < 2 * D_IN) ? W1r[(k - D_IN) * HIDDEN + col]
                                 : 0.0f;
        wc1[col * 128 + k] = (_Float16)v;
    } else if (tid < 128 * 128 + 16 * 128) {
        int t = tid - 128 * 128;
        int col = t >> 7, k = t & 127;
        float v = (col < NCLS) ? W2l[k * NCLS + col] : W2r[k * NCLS + (col - NCLS)];
        wc2[col * 128 + k] = (_Float16)v;
    }
}

// Edge pass 1: agg1[dst] += x[src] (stride-64 rows), deg[dst] += 1. 64 lanes/edge.
__global__ void scatter_x_k(const float* __restrict__ x, const int* __restrict__ src,
                            const int* __restrict__ dst, float* __restrict__ agg1,
                            float* __restrict__ deg) {
    unsigned tid = blockIdx.x * blockDim.x + threadIdx.x;
    unsigned e = tid >> 6, d = tid & 63;
    int s = src[e], t = dst[e];
    if (d < D_IN) atomicAdd(&agg1[(size_t)t * 64 + d], x[(size_t)s * D_IN + d]);
    if (d == 0)   atomicAdd(&deg[t], 1.0f);
}

// Pack per-node concatenated features [agg/deg (50) , x (50) , 0-pad (28)] as f16,
// row-major stride 128, so gemm1 A fragments are two contiguous 16B loads.
__global__ void prep_feat_k(const float* __restrict__ x, const float* __restrict__ agg1,
                            const float* __restrict__ deg, _Float16* __restrict__ fc) {
    unsigned tid = blockIdx.x * blockDim.x + threadIdx.x;  // < N_NODES*128
    unsigned n = tid >> 7, k = tid & 127;
    float invdeg = 1.0f / fmaxf(deg[n], 1.0f);
    float v = (k < D_IN)     ? agg1[(size_t)n * 64 + k] * invdeg
            : (k < 2 * D_IN) ? x[(size_t)n * D_IN + (k - D_IN)]
                             : 0.0f;
    fc[tid] = (_Float16)v;
}

// Layer 1: h = relu(fc @ [W1_l;W1_r] + b1), stored as f16.
// One wave per 16-row tile; 8 WMMA accumulators cover all 128 hidden cols.
__global__ void gemm1_k(const _Float16* __restrict__ fc, const _Float16* __restrict__ wc,
                        const float* __restrict__ b1, _Float16* __restrict__ h16) {
    int lane = threadIdx.x & 31;
    int mt = blockIdx.x * (blockDim.x >> 5) + (threadIdx.x >> 5);
    if (mt >= MTILES) return;                       // uniform per wave
    int m0 = mt << 4;
    int mrow = m0 + (lane & 15);
    int half = lane >> 4;

    v8f acc[8] = {};
#pragma unroll
    for (int kt = 0; kt < 4; ++kt) {
        int kb = kt * 32;
        const _Float16* ap = fc + (size_t)mrow * 128 + kb + half * 8;
        v8h alo = *(const v8h*)ap;
        v8h ahi = *(const v8h*)(ap + 16);
        v16h a;
#pragma unroll
        for (int i = 0; i < 8; ++i) { a[i] = alo[i]; a[i + 8] = ahi[i]; }
#pragma unroll
        for (int nt = 0; nt < 8; ++nt) {
            int col = nt * 16 + (lane & 15);
            const _Float16* bp = wc + col * 128 + kb + half * 8;
            v8h blo = *(const v8h*)bp;
            v8h bhi = *(const v8h*)(bp + 16);
            v16h b;
#pragma unroll
            for (int i = 0; i < 8; ++i) { b[i] = blo[i]; b[i + 8] = bhi[i]; }
            acc[nt] = __builtin_amdgcn_wmma_f32_16x16x32_f16(
                false, a, false, b, (short)0, acc[nt], false, false);
        }
    }
#pragma unroll
    for (int nt = 0; nt < 8; ++nt) {
        int col = nt * 16 + (lane & 15);
        float bias = b1[col];
#pragma unroll
        for (int j = 0; j < 8; ++j) {
            int r = m0 + j + half * 8;
            float u = acc[nt][j] + bias;
            h16[(size_t)r * 128 + col] = (_Float16)(u > 0.0f ? u : 0.0f);
        }
    }
}

// Layer 2 projection: q = h @ [W2_l | W2_r]  (N=16 = one WMMA tile, K=128).
__global__ void gemm2_k(const _Float16* __restrict__ h16, const _Float16* __restrict__ w2,
                        float* __restrict__ q) {
    int lane = threadIdx.x & 31;
    int mt = blockIdx.x * (blockDim.x >> 5) + (threadIdx.x >> 5);
    if (mt >= MTILES) return;
    int m0 = mt << 4;
    int mrow = m0 + (lane & 15);
    int half = lane >> 4;
    int col = lane & 15;

    v8f acc = {};
#pragma unroll
    for (int kt = 0; kt < 4; ++kt) {
        int kb = kt * 32;
        const _Float16* ap = h16 + (size_t)mrow * 128 + kb + half * 8;
        v8h alo = *(const v8h*)ap;
        v8h ahi = *(const v8h*)(ap + 16);
        const _Float16* bp = w2 + col * 128 + kb + half * 8;
        v8h blo = *(const v8h*)bp;
        v8h bhi = *(const v8h*)(bp + 16);
        v16h a, b;
#pragma unroll
        for (int i = 0; i < 8; ++i) {
            a[i] = alo[i]; a[i + 8] = ahi[i];
            b[i] = blo[i]; b[i + 8] = bhi[i];
        }
        acc = __builtin_amdgcn_wmma_f32_16x16x32_f16(
            false, a, false, b, (short)0, acc, false, false);
    }
#pragma unroll
    for (int j = 0; j < 8; ++j)
        q[(size_t)(m0 + j + half * 8) * 16 + col] = acc[j];
}

// Edge pass 2 on the PROJECTED features (8 floats/edge instead of 128).
__global__ void scatter_p_k(const float* __restrict__ q, const int* __restrict__ src,
                            const int* __restrict__ dst, float* __restrict__ aggp) {
    unsigned tid = blockIdx.x * blockDim.x + threadIdx.x;
    unsigned e = tid >> 3, c = tid & 7;
    atomicAdd(&aggp[(size_t)dst[e] * 8 + c], q[(size_t)src[e] * 16 + c]);
}

// out = aggp/deg + q[:,8:16] + b2
__global__ void final_k(const float* __restrict__ aggp, const float* __restrict__ q,
                        const float* __restrict__ deg, const float* __restrict__ b2,
                        float* __restrict__ out) {
    unsigned tid = blockIdx.x * blockDim.x + threadIdx.x;
    if (tid >= (unsigned)(N_NODES * NCLS)) return;
    unsigned n = tid >> 3, c = tid & 7;
    float invdeg = 1.0f / fmaxf(deg[n], 1.0f);
    out[tid] = aggp[tid] * invdeg + q[(size_t)n * 16 + 8 + c] + b2[c];
}

extern "C" void kernel_launch(void* const* d_in, const int* in_sizes, int n_in,
                              void* d_out, int out_size, void* d_ws, size_t ws_size,
                              hipStream_t stream) {
    const float* x   = (const float*)d_in[0];
    const int*   ei  = (const int*)d_in[1];
    const int*   src = ei;
    const int*   dst = ei + N_EDGES;
    const float* W1l = (const float*)d_in[2];
    const float* W1r = (const float*)d_in[3];
    const float* b1  = (const float*)d_in[4];
    const float* W2l = (const float*)d_in[5];
    const float* W2r = (const float*)d_in[6];
    const float* b2  = (const float*)d_in[7];
    float* out = (float*)d_out;

    char* ws = (char*)d_ws;
    float*    deg  = (float*)(ws + OFF_DEG);
    float*    agg1 = (float*)(ws + OFF_AGG1);
    _Float16* h16  = (_Float16*)(ws + OFF_H16);
    float*    q    = (float*)(ws + OFF_Q);
    float*    aggp = (float*)(ws + OFF_AGGP);
    _Float16* wc1  = (_Float16*)(ws + OFF_WC1);
    _Float16* wc2  = (_Float16*)(ws + OFF_WC2);
    _Float16* fc16 = (_Float16*)(ws + OFF_FC16);

    // zero accumulators (deg + agg1 are contiguous; aggp separate)
    (void)hipMemsetAsync(ws, 0, OFF_H16, stream);
    (void)hipMemsetAsync(ws + OFF_AGGP, 0, (size_t)N_NODES * 8 * sizeof(float), stream);

    prep_w_k<<<72, 256, 0, stream>>>(W1l, W1r, W2l, W2r, wc1, wc2);
    scatter_x_k<<<(N_EDGES * 64) / 256, 256, 0, stream>>>(x, src, dst, agg1, deg);
    prep_feat_k<<<(N_NODES * 128) / 256, 256, 0, stream>>>(x, agg1, deg, fc16);
    gemm1_k<<<(MTILES + 7) / 8, 256, 0, stream>>>(fc16, wc1, b1, h16);
    gemm2_k<<<(MTILES + 7) / 8, 256, 0, stream>>>(h16, wc2, q);
    scatter_p_k<<<(N_EDGES * 8) / 256, 256, 0, stream>>>(q, src, dst, aggp);
    final_k<<<(N_NODES * NCLS) / 256, 256, 0, stream>>>(aggp, q, deg, b2, out);
}
